// HybridGNN_50766513439446
// MI455X (gfx1250) — matrix operand order repaired and checked by reading, not compiled, not run on hardware
//
#include <hip/hip_runtime.h>

typedef __attribute__((ext_vector_type(16))) _Float16 v16h;
typedef __attribute__((ext_vector_type(8)))  _Float16 v8h;
typedef __attribute__((ext_vector_type(8)))  float    v8f;

#define BB 64     // batch
#define TT 64     // seq len
#define NN 256    // agents per graph
#define FF 16     // agent feature dim
#define HH 64     // gnn hidden
#define GH 256    // 4*H (stacked gates)

#define LOG2E  1.4426950408889634f

union HFrag { v16h v; v8h h[2]; };

// TRANS-unit activations: v_exp_f32 + v_rcp_f32, no IEEE divide sequences.
__device__ __forceinline__ float fsig(float x) {
    return __builtin_amdgcn_rcpf(1.0f + __builtin_amdgcn_exp2f(-LOG2E * x));
}
__device__ __forceinline__ float ftanh(float x) {
    return __builtin_fmaf(2.0f,
        __builtin_amdgcn_rcpf(1.0f + __builtin_amdgcn_exp2f(-2.0f * LOG2E * x)),
        -1.0f);
}

// ---------------------------------------------------------------------------
// Persistent GCLSTM scan: each wave owns 16 nodes, runs all 64 timesteps with
// weights (f16) and recurrent h (f16) resident in LDS; all GEMMs via
// v_wmma_f32_16x16x32_f16 (f32 accumulate). Mean-pool via atomics into d_ws.
// ---------------------------------------------------------------------------
__global__ __launch_bounds__(128)
void gclstm_wmma_kernel(const float* __restrict__ agent_obs,   // [B,T,N,F]
                        const float* __restrict__ Wx,          // [F,4H]
                        const float* __restrict__ Wh,          // [H,4H]
                        const float* __restrict__ b_conv,      // [4H]
                        const float* __restrict__ b_gate,      // [4H]
                        const float* __restrict__ w_peep,      // [3,H]
                        float* __restrict__ res_sum)           // [B,H] pooled sums
{
    // LDS-resident transposed f16 weights in WMMA B-fragment-friendly order.
    __shared__ __align__(32) _Float16 wxT[GH * 32];      // [n][p], k = p with bits 3,4 swapped (table order), padded 16->32
    __shared__ __align__(32) _Float16 whT[GH * 64];      // [n][k]
    __shared__ float ball[GH];                           // b_conv + b_gate
    __shared__ __align__(32) _Float16 hbuf[4 * 16 * 64]; // per-wave h state, row-major [node][k]

    const int tid = threadIdx.x;

    for (int i = tid; i < GH * 32; i += 128) {
        int n = i >> 5, p = i & 31, k = (p & 7) | ((p & 8) << 1) | ((p & 16) >> 1);
        wxT[i] = (k < FF) ? (_Float16)Wx[k * GH + n] : (_Float16)0.0f;
    }
    for (int i = tid; i < GH * 64; i += 128) {
        int n = i >> 6, k = i & 63;
        whT[i] = (_Float16)Wh[k * GH + n];
    }
    for (int i = tid; i < GH; i += 128) ball[i] = b_conv[i] + b_gate[i];
    for (int i = tid; i < 4 * 16 * 64; i += 128) hbuf[i] = (_Float16)0.0f;
    __syncthreads();

    const int wave = tid >> 5;
    const int lane = tid & 31;
    const int half = lane >> 4;   // lane group: rows 0..7 vs 8..15 in C/D layout
    const int mrow = lane & 15;

    const int m0 = blockIdx.x * 64 + wave * 16;   // first node of this wave's tile
    const int b  = m0 / NN;
    const int n0 = m0 % NN;

    _Float16* hw = &hbuf[wave * 16 * 64];

    // peephole scalars: per-lane column j = 16*jc + mrow
    float p0[4], p1[4], p2[4];
#pragma unroll
    for (int jc = 0; jc < 4; ++jc) {
        int j = jc * 16 + mrow;
        p0[jc] = w_peep[0 * HH + j];
        p1[jc] = w_peep[1 * HH + j];
        p2[jc] = w_peep[2 * HH + j];
    }

    v8f cst[4];      // cell state, C/D layout: elem v -> node (v + 8*half), col 16*jc+mrow
    v8f hlast[4];    // last hidden (f32, for exact pooling)
#pragma unroll
    for (int jc = 0; jc < 4; ++jc)
#pragma unroll
        for (int e = 0; e < 8; ++e) { cst[jc][e] = 0.0f; hlast[jc][e] = 0.0f; }

    const float* xbase = agent_obs + (size_t)b * TT * NN * FF;

#pragma unroll 1
    for (int t = 0; t < TT; ++t) {
        // ---- x_t A-fragment (16x32, K=16..31 zero-padded) -------------------
        const float* xr = xbase + ((size_t)t * NN + (n0 + mrow)) * FF + half * 8;
        v16h ax;
#pragma unroll
        for (int e = 0; e < 8; ++e) ax[e] = (_Float16)xr[e];
#pragma unroll
        for (int e = 8; e < 16; ++e) ax[e] = (_Float16)0.0f;
        if (t + 1 < TT) __builtin_prefetch(xr + NN * FF, 0, 3);  // global_prefetch_b8

        // ---- h A-fragments from LDS (K=0..31, K=32..63) --------------------
        HFrag ah0, ah1;
        {
            const _Float16* hp = hw + mrow * 64 + half * 16;
            ah0.h[0] = *(const v8h*)(hp);        // K 0..7   / 16..23
            ah0.h[1] = *(const v8h*)(hp + 8);    // K 8..15  / 24..31
            ah1.h[0] = *(const v8h*)(hp + 32);   // K 32..39 / 48..55
            ah1.h[1] = *(const v8h*)(hp + 40);   // K 40..47 / 56..63
        }

#pragma unroll
        for (int jc = 0; jc < 4; ++jc) {
            v8f acc[4];
#pragma unroll
            for (int g = 0; g < 4; ++g) {        // g: 0=i,1=f,2=c,3=o
                const int chunk = g * 4 + jc;    // output column chunk
                const int col   = chunk * 16 + mrow;
                const float bias = ball[col];
                v8f a;
#pragma unroll
                for (int e = 0; e < 8; ++e) a[e] = bias;
                v16h bwx = *(const v16h*)&wxT[col * 32 + half * 16];
                v16h bh0 = *(const v16h*)&whT[col * 64 + half * 16];
                v16h bh1 = *(const v16h*)&whT[col * 64 + 32 + half * 16];
                a = __builtin_amdgcn_wmma_f32_16x16x32_f16(false, ax,    false, bwx, (short)0, a, false, false);
                a = __builtin_amdgcn_wmma_f32_16x16x32_f16(false, ah0.v, false, bh0, (short)0, a, false, false);
                a = __builtin_amdgcn_wmma_f32_16x16x32_f16(false, ah1.v, false, bh1, (short)0, a, false, false);
                acc[g] = a;
            }

            // ---- elementwise peephole-LSTM update in C/D layout ------------
            v8f cj = cst[jc];
            v8f hn;
#pragma unroll
            for (int e = 0; e < 8; ++e) {
                float ig = fsig(acc[0][e] + p0[jc] * cj[e]);
                float fg = fsig(acc[1][e] + p1[jc] * cj[e]);
                float gg = ftanh(acc[2][e]);
                float cn = fg * cj[e] + ig * gg;
                float og = fsig(acc[3][e] + p2[jc] * cn);   // peephole on NEW cell
                cj[e] = cn;
                hn[e] = og * ftanh(cn);
            }
            cst[jc]   = cj;
            hlast[jc] = hn;

            // write h (f16) back to LDS for next step's A-fragments
            const int j = jc * 16 + mrow;
#pragma unroll
            for (int e = 0; e < 8; ++e)
                hw[(e + 8 * half) * 64 + j] = (_Float16)hn[e];
        }
    }

    // ---- global mean pool (sum here; divide in head kernel) ----------------
#pragma unroll
    for (int jc = 0; jc < 4; ++jc) {
        float s = 0.0f;
#pragma unroll
        for (int e = 0; e < 8; ++e) s += hlast[jc][e];
        atomicAdd(&res_sum[b * HH + jc * 16 + mrow], s);
    }
}

// ---------------------------------------------------------------------------
// Head: out = tanh(mean(h)@Wl_a + b_a + hideout@Wl_h + b_h).  (Wr_* terms
// multiply an all-zero x_ss and vanish.)  4096 outputs -> trivially small.
// ---------------------------------------------------------------------------
__global__ __launch_bounds__(256)
void head_kernel(const float* __restrict__ res_sum,   // [B,H] sums over 256 nodes
                 const float* __restrict__ hideout,   // [B,2]
                 const float* __restrict__ Wl_a,      // [H,O]
                 const float* __restrict__ b_a,       // [O]
                 const float* __restrict__ Wl_h,      // [2,O]
                 const float* __restrict__ b_h,       // [O]
                 float* __restrict__ out)             // [B,O]
{
    const int idx = blockIdx.x * blockDim.x + threadIdx.x;
    if (idx >= BB * HH) return;
    const int b = idx >> 6;
    const int o = idx & 63;

    float acc = b_a[o] + b_h[o];
    acc += hideout[b * 2 + 0] * Wl_h[0 * HH + o];
    acc += hideout[b * 2 + 1] * Wl_h[1 * HH + o];
    const float inv = 1.0f / (float)NN;
#pragma unroll 8
    for (int h = 0; h < HH; ++h)
        acc += (res_sum[b * HH + h] * inv) * Wl_a[h * HH + o];
    out[idx] = ftanh(acc);
}

extern "C" void kernel_launch(void* const* d_in, const int* in_sizes, int n_in,
                              void* d_out, int out_size, void* d_ws, size_t ws_size,
                              hipStream_t stream) {
    const float* agent_obs = (const float*)d_in[0];
    const float* hideout   = (const float*)d_in[1];
    // d_in[2] timestep_obs: unused by the reference math
    const float* Wx     = (const float*)d_in[3];
    const float* Wh     = (const float*)d_in[4];
    const float* b_conv = (const float*)d_in[5];
    const float* b_gate = (const float*)d_in[6];
    const float* w_peep = (const float*)d_in[7];
    const float* Wl_a   = (const float*)d_in[8];
    // d_in[9] Wr_a: multiplies zeros
    const float* b_a    = (const float*)d_in[10];
    const float* Wl_h   = (const float*)d_in[11];
    // d_in[12] Wr_h: multiplies zeros
    const float* b_h    = (const float*)d_in[13];

    float* out     = (float*)d_out;
    float* res_sum = (float*)d_ws;   // [B*H] f32 accumulator

    (void)hipMemsetAsync(res_sum, 0, BB * HH * sizeof(float), stream);

    // 16384 nodes / 64 per block (4 waves x 16 nodes) = 256 blocks
    gclstm_wmma_kernel<<<256, 128, 0, stream>>>(agent_obs, Wx, Wh, b_conv, b_gate,
                                                w_peep, res_sum);
    head_kernel<<<16, 256, 0, stream>>>(res_sum, hideout, Wl_a, b_a, Wl_h, b_h, out);
}
